// PTAttentionHead_8151847927835
// MI455X (gfx1250) — compile-verified, matrix operand order checked by reading, not run on hardware
//
#include <hip/hip_runtime.h>

// ---------------- problem constants ----------------
#define BB 2048
#define TT 128
#define CC 192
#define HH 96
#define NC 38
#define SS 25
#define H3 288          // 3*H
#define KX 256          // padded K for GRU input GEMM (C+NC=230 -> 256)
#define NG 64           // padded N for generator GEMM (38 -> 64, two 16-wide tiles per wave)

typedef float v8f   __attribute__((ext_vector_type(8)));
typedef __bf16 v16bf __attribute__((ext_vector_type(16)));
typedef __bf16 bf16x8 __attribute__((ext_vector_type(8)));

// CDNA5 hardware transcendental tanh (TRANS32 op, co-issues with VALU).
__device__ __forceinline__ float htanh(float x) {
    float r;
    asm volatile("v_tanh_f32 %0, %1" : "=v"(r) : "v"(x));
    return r;
}
// sigmoid from hw tanh: one TRANS op instead of exp+add+rcp
__device__ __forceinline__ float hsigmoid(float x) {
    return 0.5f * (1.0f + htanh(0.5f * x));
}

// ============================================================
// bf16 GEMM:  out[M,N] = A[M,K](bf16) * B[N,K]^T(bf16) + bias
// Block = 256 threads (8 waves). Each wave -> 16 rows x 32 cols
// (two 16x16 WMMA tiles, A-fragment reused). The 32xK B-panel is
// staged into LDS once per block with global_load_async_to_lds_b128
// (ASYNCcnt), then consumed via ds_load_b128.
// Grid: (M/128, N/32). K multiple of 32, K <= 256, ldb == K.
// ============================================================
__global__ __launch_bounds__(256) void gemm_bf16_wmma(
    const __bf16* __restrict__ A, int lda,
    const __bf16* __restrict__ Bm, int ldb,
    const float* __restrict__ bias,
    float* __restrict__ outF, __bf16* __restrict__ outB, int ldo,
    int K, int n_valid)
{
    __shared__ __bf16 Bs[32 * 256];          // 16 KB B panel

    const int tid  = threadIdx.x;
    const int wave = tid >> 5;
    const int lane = tid & 31;
    const int half = lane >> 4;              // 0: lanes 0-15, 1: lanes 16-31
    const int l15  = lane & 15;

    const int m0  = (blockIdx.x * 8 + wave) * 16;
    const int n0b = blockIdx.y * 32;

    // ---- async-stage B panel (rows n0b..n0b+31, all K) into LDS ----
    {
        const int kc8    = K >> 3;           // 8-elem (16B) chunks per row
        const int chunks = 32 * kc8;
        const unsigned lbase = (unsigned)(uintptr_t)(&Bs[0]);  // LDS aperture: addr[31:0] = LDS offset
        for (int c = tid; c < chunks; c += 256) {
            int row = c / kc8, cc = c - row * kc8;
            const __bf16* g = Bm + (size_t)(n0b + row) * ldb + cc * 8;
            unsigned l = lbase + (unsigned)c * 16u;
            asm volatile("global_load_async_to_lds_b128 %0, %1, off"
                         :: "v"(l), "v"(g) : "memory");
        }
        asm volatile("s_wait_asynccnt 0" ::: "memory");
    }
    __syncthreads();

    // Per-lane A base per the ISA 16-bit A(16x32) layout:
    // lane<16 : row m0+l15, K chunks [0..7] and [16..23]
    // lane>=16: row m0+l15, K chunks [8..15] and [24..31]
    const __bf16* Ap  = A + (size_t)(m0 + l15) * lda + half * 8;
    const __bf16* Bs0 = Bs + (size_t)l15 * K + half * 8;        // n-tile 0 row
    const __bf16* Bs1 = Bs0 + (size_t)16 * K;                   // n-tile 1 row

    v8f c0 = {}, c1 = {};
    for (int ks = 0; ks < K; ks += 32) {
        bf16x8 alo = *(const bf16x8*)(Ap + ks);
        bf16x8 ahi = *(const bf16x8*)(Ap + ks + 16);
        bf16x8 b0l = *(const bf16x8*)(Bs0 + ks);
        bf16x8 b0h = *(const bf16x8*)(Bs0 + ks + 16);
        bf16x8 b1l = *(const bf16x8*)(Bs1 + ks);
        bf16x8 b1h = *(const bf16x8*)(Bs1 + ks + 16);
        v16bf av, bv0, bv1;
#pragma unroll
        for (int i = 0; i < 8; ++i) {
            av[i]  = alo[i]; av[i + 8]  = ahi[i];
            bv0[i] = b0l[i]; bv0[i + 8] = b0h[i];
            bv1[i] = b1l[i]; bv1[i + 8] = b1h[i];
        }
        c0 = __builtin_amdgcn_wmma_f32_16x16x32_bf16(
                 false, av, false, bv0, (short)0, c0, false, false);
        c1 = __builtin_amdgcn_wmma_f32_16x16x32_bf16(
                 false, av, false, bv1, (short)0, c1, false, false);
    }

    // D layout: lane -> n = ntile_base + l15 ; rows m0 + half*8 + i
    const int mbase = m0 + half * 8;
#pragma unroll
    for (int j = 0; j < 2; ++j) {
        const v8f& c = j ? c1 : c0;
        const int n = n0b + j * 16 + l15;
        if (n < n_valid) {
            const float bb = bias ? bias[n] : 0.0f;
#pragma unroll
            for (int i = 0; i < 8; ++i) {
                float v = c[i] + bb;
                size_t idx = (size_t)(mbase + i) * ldo + n;
                if (outF) outF[idx] = v;
                if (outB) outB[idx] = (__bf16)v;
            }
        }
    }
}

// ============================================================
// f32 -> bf16 conversion with row/col zero padding
// ============================================================
__global__ __launch_bounds__(256) void cvt_pad_kernel(
    const float* __restrict__ src, __bf16* __restrict__ dst,
    int srows, int scols, int rows, int dcols)
{
    int i = blockIdx.x * blockDim.x + threadIdx.x;
    if (i >= rows * dcols) return;
    int r = i / dcols, c = i - r * dcols;
    float v = (r < srows && c < scols) ? src[r * scols + c] : 0.0f;
    dst[i] = (__bf16)v;
}

__global__ __launch_bounds__(256) void init_h_kernel(
    float* __restrict__ h, __bf16* __restrict__ hbf)
{
    int i = blockIdx.x * blockDim.x + threadIdx.x;
    if (i < BB * HH) { h[i] = 0.0f; hbf[i] = (__bf16)0.0f; }
}

// ============================================================
// Fused attention step: per block = one batch row b.
//   e[t]   = sum_h tanh(Hproj[b,t,h] + hp[b,h]) * sv[h]   (hw v_tanh_f32)
//   alpha  = softmax_t(e)
//   ctx[c] = sum_t alpha[t] * X[b,t,c]
//   xcat[b] = [ctx (192) | onehot(target[b,s]) (38) | zeros] (bf16, 256)
// ============================================================
__global__ __launch_bounds__(256) void attn_kernel(
    const __bf16* __restrict__ Hproj, const __bf16* __restrict__ Xbf,
    const float* __restrict__ hp, const float* __restrict__ w_score,
    const int* __restrict__ targets, int s,
    __bf16* __restrict__ xcat)
{
    __shared__ float hp_s[HH], sv_s[HH], e_s[TT], red[TT];
    const int b = blockIdx.x, tid = threadIdx.x;

    if (tid < HH) { hp_s[tid] = hp[b * HH + tid]; sv_s[tid] = w_score[tid]; }
    __syncthreads();

    float ev = -1e30f;
    if (tid < TT) {
        const bf16x8* row = (const bf16x8*)(Hproj + ((size_t)b * TT + tid) * HH);
        float acc = 0.0f;
#pragma unroll
        for (int j = 0; j < HH / 8; ++j) {
            bf16x8 v = row[j];
#pragma unroll
            for (int k = 0; k < 8; ++k) {
                int h = j * 8 + k;
                acc += htanh((float)v[k] + hp_s[h]) * sv_s[h];
            }
        }
        ev = acc;
    }
    // softmax over T=128
    if (tid < TT) red[tid] = ev;
    __syncthreads();
    for (int off = TT / 2; off > 0; off >>= 1) {
        if (tid < off) red[tid] = fmaxf(red[tid], red[tid + off]);
        __syncthreads();
    }
    const float mx = red[0];
    __syncthreads();
    float ex = 0.0f;
    if (tid < TT) { ex = __expf(ev - mx); red[tid] = ex; }
    __syncthreads();
    for (int off = TT / 2; off > 0; off >>= 1) {
        if (tid < off) red[tid] += red[tid + off];
        __syncthreads();
    }
    const float inv = 1.0f / red[0];
    if (tid < TT) e_s[tid] = ex * inv;     // alpha
    __syncthreads();

    if (tid < CC) {
        // coalesced across threads: for fixed t, threads read contiguous row
        const __bf16* xb = Xbf + (size_t)b * TT * CC + tid;
        float acc = 0.0f;
#pragma unroll 4
        for (int t = 0; t < TT; ++t) acc += e_s[t] * (float)xb[(size_t)t * CC];
        xcat[(size_t)b * KX + tid] = (__bf16)acc;
    } else {
        int j = tid - CC;                  // 0..63 covers onehot + zero pad
        int tgt = targets[b * SS + s];
        float v = (j < NC && j == tgt) ? 1.0f : 0.0f;
        xcat[(size_t)b * KX + CC + j] = (__bf16)v;
    }
}

// ============================================================
// GRU elementwise update (torch semantics) with hw tanh/sigmoid.
// Writes f32 h, bf16 h, and bf16 history in [B,S,H] order.
// ============================================================
__global__ __launch_bounds__(256) void gru_kernel(
    const float* __restrict__ gi, const float* __restrict__ gh,
    float* __restrict__ h, __bf16* __restrict__ hbf,
    __bf16* __restrict__ Hs, int s)
{
    int i = blockIdx.x * blockDim.x + threadIdx.x;
    if (i >= BB * HH) return;
    int b = i / HH, k = i - b * HH;
    const float* gib = gi + (size_t)b * H3;
    const float* ghb = gh + (size_t)b * H3;
    float r = hsigmoid(gib[k]        + ghb[k]);
    float z = hsigmoid(gib[HH + k]   + ghb[HH + k]);
    float n = htanh(gib[2 * HH + k] + r * ghb[2 * HH + k]);
    float hn = (1.0f - z) * n + z * h[i];
    h[i] = hn;
    hbf[i] = (__bf16)hn;
    Hs[((size_t)b * SS + s) * HH + k] = (__bf16)hn;
}

// ============================================================
extern "C" void kernel_launch(void* const* d_in, const int* in_sizes, int n_in,
                              void* d_out, int out_size, void* d_ws, size_t ws_size,
                              hipStream_t stream)
{
    const float* inputs  = (const float*)d_in[0];
    const int*   targets = (const int*)d_in[1];
    const float* w_i2h   = (const float*)d_in[2];
    const float* w_h2h   = (const float*)d_in[3];
    const float* b_h2h   = (const float*)d_in[4];
    const float* w_score = (const float*)d_in[5];
    const float* w_ih    = (const float*)d_in[6];
    const float* w_hh    = (const float*)d_in[7];
    const float* b_ih    = (const float*)d_in[8];
    const float* b_hh    = (const float*)d_in[9];
    const float* w_gen   = (const float*)d_in[10];
    const float* b_gen   = (const float*)d_in[11];
    float* probs = (float*)d_out;

    // ---------------- workspace carve-up (256B aligned) ----------------
    char* ws = (char*)d_ws;
    size_t off = 0;
    auto carve = [&](size_t bytes) -> char* {
        off = (off + 255) & ~(size_t)255;
        char* p = ws + off; off += bytes; return p;
    };
    __bf16* Xbf    = (__bf16*)carve((size_t)BB * TT * CC * 2);   // 100.7 MB (L2-resident)
    __bf16* Hproj  = (__bf16*)carve((size_t)BB * TT * HH * 2);   //  50.3 MB (L2-resident)
    __bf16* Hs     = (__bf16*)carve((size_t)BB * SS * HH * 2);   //   9.8 MB
    float*  h_f    = (float*)carve((size_t)BB * HH * 4);
    __bf16* h_bf   = (__bf16*)carve((size_t)BB * HH * 2);
    float*  hp_f   = (float*)carve((size_t)BB * HH * 4);
    __bf16* xcat   = (__bf16*)carve((size_t)BB * KX * 2);
    float*  gi_f   = (float*)carve((size_t)BB * H3 * 4);
    float*  gh_f   = (float*)carve((size_t)BB * H3 * 4);
    __bf16* Wi2h   = (__bf16*)carve((size_t)HH * CC * 2);
    __bf16* Wh2h   = (__bf16*)carve((size_t)HH * HH * 2);
    __bf16* Wih    = (__bf16*)carve((size_t)H3 * KX * 2);
    __bf16* Whh    = (__bf16*)carve((size_t)H3 * HH * 2);
    __bf16* Wgen   = (__bf16*)carve((size_t)NG * HH * 2);
    (void)ws_size; (void)in_sizes; (void)n_in; (void)out_size;

    auto cvt = [&](const float* s, __bf16* d, int srows, int scols, int rows, int dcols) {
        int n = rows * dcols;
        cvt_pad_kernel<<<(n + 255) / 256, 256, 0, stream>>>(s, d, srows, scols, rows, dcols);
    };
    auto gemm = [&](const __bf16* A, int lda, const __bf16* Bm, int ldb,
                    const float* bias, float* oF, __bf16* oB, int ldo,
                    int M, int N, int K, int n_valid) {
        dim3 g(M / 128, N / 32);
        gemm_bf16_wmma<<<g, 256, 0, stream>>>(A, lda, Bm, ldb, bias, oF, oB, ldo, K, n_valid);
    };

    // ---------------- one-time conversions ----------------
    cvt(inputs, Xbf, BB * TT, CC, BB * TT, CC);       // X -> bf16 (halves L2 footprint)
    cvt(w_i2h, Wi2h, HH, CC, HH, CC);
    cvt(w_h2h, Wh2h, HH, HH, HH, HH);
    cvt(w_ih,  Wih,  H3, CC + NC, H3, KX);            // pad K 230 -> 256 with zeros
    cvt(w_hh,  Whh,  H3, HH, H3, HH);
    cvt(w_gen, Wgen, NC, HH, NG, HH);                 // pad N 38 -> 64 with zeros
    init_h_kernel<<<(BB * HH + 255) / 256, 256, 0, stream>>>(h_f, h_bf);

    // ---------------- Hproj = X * W_i2h^T  (bf16 out, stays in L2) ----------------
    gemm(Xbf, CC, Wi2h, CC, nullptr, nullptr, Hproj, HH, BB * TT, HH, CC, HH);

    // ---------------- recurrence ----------------
    for (int s = 0; s < SS; ++s) {
        // hp = h * W_h2h^T + b_h2h
        gemm(h_bf, HH, Wh2h, HH, b_h2h, hp_f, nullptr, HH, BB, HH, HH, HH);
        // fused score / softmax / context / onehot
        attn_kernel<<<BB, 256, 0, stream>>>(Hproj, Xbf, hp_f, w_score, targets, s, xcat);
        // GRU gates
        gemm(xcat, KX, Wih, KX, b_ih, gi_f, nullptr, H3, BB, H3, KX, H3);
        gemm(h_bf, HH, Whh, HH, b_hh, gh_f, nullptr, H3, BB, H3, HH, H3);
        // GRU update (+ history write in [B,S,H] order)
        gru_kernel<<<(BB * HH + 255) / 256, 256, 0, stream>>>(gi_f, gh_f, h_f, h_bf, Hs, s);
    }

    // ---------------- probs = Hs * W_gen^T + b_gen  (direct to d_out [B,S,NC]) ----
    gemm(Hs, HH, Wgen, HH, b_gen, probs, nullptr, NC, BB * SS, NG, HH, NC);
}
